// MGCRN_86191403696618
// MI455X (gfx1250) — matrix operand order repaired and checked by reading, not compiled, not run on hardware
//
#include <hip/hip_runtime.h>
#include <math.h>

// ---------------- problem constants ----------------
#define NN    2048
#define Bb    32
#define Tt    12
#define HZN_  12
#define D_IN_ 2
#define RU_   64
#define EMB_  8
#define GIN_  66            // D_IN + RU  ==  D_OUT + YC + RU
#define BC_   (Bb * GIN_)   // 2112 columns of the supports GEMM
#define FPAD_ 416           // 6*66 = 396 padded up to multiple of 32
#define FREAL 396

// ---------------- GEMM tile config ----------------
#define BM 128
#define BN 64
#define BK 32
#define LDS_S 40            // LDS row stride (halves): 80B, 16B-aligned, bank-skewed

typedef unsigned short u16;
typedef unsigned int   u32;
typedef __attribute__((ext_vector_type(16))) __bf16 v16bf;
typedef __attribute__((ext_vector_type(8)))  float  v8f;

union AFrag { v16bf v; u32 u[8]; };
union CFrag { v8f   v; float f[8]; };

// epilogue modes
#define MODE_SCATTER   0   // bf16 scatter into XG feature slot
#define MODE_CHEB      1   // bf16 out = 2*acc - I
#define MODE_BIAS_SIG  2   // f32 out = sigmoid(acc + bias)
#define MODE_BIAS_TANH 3   // f32 out = tanh(acc + bias)

__device__ __forceinline__ u16 f2bf(float f) {
    union { float f; u32 u; } x; x.f = f;
    u32 r = x.u + 0x7FFFu + ((x.u >> 16) & 1u);   // round-to-nearest-even
    return (u16)(r >> 16);
}

// ------------------------------------------------------------------
// bf16 WMMA GEMM: C(MxNcols) = A(MxK) * B(KxNcols), fused epilogues.
// A row-major bf16 (lda), B row-major bf16 (ldb). M mult of 128,
// Ncols mult of 64, K mult of 32. 256 threads = 8 waves; each wave
// owns a 32x32 output patch (2x2 WMMA tiles).
// ------------------------------------------------------------------
__global__ __launch_bounds__(256) void wmma_gemm_bf16(
    const u16* __restrict__ A, long lda,
    const u16* __restrict__ B, long ldb,
    int Kdim,
    u16*   __restrict__ out_bf,
    float* __restrict__ out_f,
    long ldc,
    const float* __restrict__ bias,
    int mode, int kslot)
{
    __shared__ u16 As[BM * LDS_S];
    __shared__ u16 Bs[BN * LDS_S];

    const int tid  = threadIdx.x;
    const int lane = tid & 31;
    const int wid  = tid >> 5;
    const int wm   = wid & 3;          // 4 wave-rows of 32
    const int wn   = wid >> 2;         // 2 wave-cols of 32
    const int half = lane >> 4;
    const int l16  = lane & 15;

    const long blockM = (long)blockIdx.y * BM;
    const long blockN = (long)blockIdx.x * BN;

    CFrag acc[2][2];
#pragma unroll
    for (int mt = 0; mt < 2; ++mt)
#pragma unroll
        for (int sub = 0; sub < 2; ++sub)
#pragma unroll
            for (int i = 0; i < 8; ++i) acc[mt][sub].f[i] = 0.0f;

    // cooperative load mapping
    const int am  = tid >> 2;          // 0..63 (A tile row, +64 for second half)
    const int ak  = (tid & 3) * 8;     // 0,8,16,24
    const int bk  = tid >> 3;          // 0..31 (B tile k-row)
    const int bn0 = (tid & 7) * 8;     // 0..56

    for (int k0 = 0; k0 < Kdim; k0 += BK) {
        // A tile: 128x32 bf16, vectorized 16B loads, K-contiguous in LDS
#pragma unroll
        for (int r = 0; r < 2; ++r) {
            uint4 av = *(const uint4*)(A + (blockM + am + r * 64) * lda + (long)(k0 + ak));
            *(uint4*)&As[(am + r * 64) * LDS_S + ak] = av;
        }
        // B tile: 32(k) x 64(n), transpose into LDS [n][k]
        uint4 bv = *(const uint4*)(B + (long)(k0 + bk) * ldb + blockN + bn0);
        const u16* bh = (const u16*)&bv;
#pragma unroll
        for (int i = 0; i < 8; ++i)
            Bs[(bn0 + i) * LDS_S + bk] = bh[i];

        // prefetch next k-tile while this one is consumed (global_prefetch_b8)
        if (k0 + BK < Kdim) {
            __builtin_prefetch(A + (blockM + am) * lda + (long)(k0 + BK + ak), 0, 0);
            __builtin_prefetch(B + (long)(k0 + BK + bk) * ldb + blockN + bn0, 0, 0);
        }
        __syncthreads();

        // A fragments (16x32): lanes 0-15 rows M, K in {0..7,16..23};
        // lanes 16-31 K in {8..15,24..31}  (ISA 7.12.2)
        AFrag a[2];
#pragma unroll
        for (int mt = 0; mt < 2; ++mt)
#pragma unroll
            for (int v = 0; v < 8; ++v) {
                int kk = ((v < 4) ? 0 : 16) + half * 8 + (v & 3) * 2;
                a[mt].u[v] = *(const u32*)&As[(wm * 32 + mt * 16 + l16) * LDS_S + kk];
            }
        // B fragments (32x16): lanes 0-15 K in {0..7,16..23}, lanes 16-31 K in {8..15,24..31}
        AFrag bf[2];
#pragma unroll
        for (int sub = 0; sub < 2; ++sub)
#pragma unroll
            for (int v = 0; v < 8; ++v) {
                int kk = ((v < 4) ? 0 : 16) + half * 8 + (v & 3) * 2;
                bf[sub].u[v] = *(const u32*)&Bs[(wn * 32 + sub * 16 + l16) * LDS_S + kk];
            }
#pragma unroll
        for (int mt = 0; mt < 2; ++mt)
#pragma unroll
            for (int sub = 0; sub < 2; ++sub)
                acc[mt][sub].v = __builtin_amdgcn_wmma_f32_16x16x32_bf16(
                    false, a[mt].v, false, bf[sub].v, (short)0, acc[mt][sub].v,
                    false, false);
        __syncthreads();
    }

    // epilogue: D layout = VGPR v holds row (v + 8*half), col = lane&15
#pragma unroll
    for (int mt = 0; mt < 2; ++mt)
#pragma unroll
    for (int sub = 0; sub < 2; ++sub) {
#pragma unroll
        for (int v = 0; v < 8; ++v) {
            long row = blockM + wm * 32 + mt * 16 + v + 8 * half;
            long col = blockN + wn * 32 + sub * 16 + l16;
            float val = acc[mt][sub].f[v];
            if (mode == MODE_SCATTER) {
                int b = (int)(col / GIN_);
                int c = (int)(col - (long)b * GIN_);
                out_bf[((long)b * NN + row) * FPAD_ + kslot * GIN_ + c] = f2bf(val);
            } else if (mode == MODE_CHEB) {
                val = 2.0f * val - ((row == col) ? 1.0f : 0.0f);
                out_bf[row * ldc + col] = f2bf(val);
            } else if (mode == MODE_BIAS_SIG) {
                val += bias[col];
                out_f[row * ldc + col] = 1.0f / (1.0f + __expf(-val));
            } else {
                val += bias[col];
                out_f[row * ldc + col] = tanhf(val);
            }
        }
    }
}

// ------------------------------------------------------------------
// adp = softmax(relu(E1 @ E2^T), axis=1) -> bf16   (one block per row)
// ------------------------------------------------------------------
__global__ __launch_bounds__(256) void adp_softmax_kernel(
    const float* __restrict__ E1, const float* __restrict__ E2,
    u16* __restrict__ adp_bf)
{
    const int row = blockIdx.x;
    const int tid = threadIdx.x;
    __shared__ float e1[EMB_];
    __shared__ float red[256];
    if (tid < EMB_) e1[tid] = E1[row * EMB_ + tid];
    __syncthreads();

    float vals[8];
    float lmax = -1e30f;
#pragma unroll
    for (int i = 0; i < 8; ++i) {
        int col = i * 256 + tid;
        float s = 0.0f;
#pragma unroll
        for (int e = 0; e < EMB_; ++e) s += e1[e] * E2[col * EMB_ + e];
        s = fmaxf(s, 0.0f);
        vals[i] = s;
        lmax = fmaxf(lmax, s);
    }
    red[tid] = lmax; __syncthreads();
    for (int s = 128; s > 0; s >>= 1) {
        if (tid < s) red[tid] = fmaxf(red[tid], red[tid + s]);
        __syncthreads();
    }
    float m = red[0]; __syncthreads();
    float lsum = 0.0f;
#pragma unroll
    for (int i = 0; i < 8; ++i) { vals[i] = __expf(vals[i] - m); lsum += vals[i]; }
    red[tid] = lsum; __syncthreads();
    for (int s = 128; s > 0; s >>= 1) {
        if (tid < s) red[tid] += red[tid + s];
        __syncthreads();
    }
    float inv = 1.0f / red[0];
#pragma unroll
    for (int i = 0; i < 8; ++i)
        adp_bf[(long)row * NN + i * 256 + tid] = f2bf(vals[i] * inv);
}

__global__ void cast_f32_bf16_kernel(const float* __restrict__ src,
                                     u16* __restrict__ dst, long n) {
    long i = (long)blockIdx.x * blockDim.x + threadIdx.x;
    if (i < n) dst[i] = f2bf(src[i]);
}

// convert weight (Kreal x Ncol f32) -> (FPAD x Ncol bf16), zero-padded rows
__global__ void prep_weight_kernel(const float* __restrict__ W,
                                   u16* __restrict__ out, int Kreal, int Ncol) {
    long i = (long)blockIdx.x * blockDim.x + threadIdx.x;
    if (i >= (long)FPAD_ * Ncol) return;
    int r = (int)(i / Ncol), c = (int)(i - (long)r * Ncol);
    out[i] = (r < Kreal) ? f2bf(W[(long)r * Ncol + c]) : (u16)0;
}

// zero the 20 pad feature columns of XG (once per launch)
__global__ void zero_pad_kernel(u16* __restrict__ XG) {
    long i = (long)blockIdx.x * blockDim.x + threadIdx.x;
    if (i >= (long)Bb * NN * (FPAD_ - FREAL)) return;
    long row = i / (FPAD_ - FREAL);
    int  c   = (int)(i - row * (FPAD_ - FREAL));
    XG[row * FPAD_ + FREAL + c] = 0;
}

// ------------------------------------------------------------------
// pack Xin (N x B*GIN bf16) = concat([xt , h or z*h]); also copies the
// identity-support slots (k=0 and k=3) of XG.
// ------------------------------------------------------------------
__global__ void pack_kernel(
    const float* __restrict__ xsrc,   // enc: x   | dec: y_cov
    const float* __restrict__ go,     // dec only
    const float* __restrict__ h,
    const float* __restrict__ zr,     // use_z: z = zr[row*128 + j]
    u16* __restrict__ Xin, u16* __restrict__ XG,
    int t, int dec, int use_z)
{
    long idx = (long)blockIdx.x * blockDim.x + threadIdx.x;
    if (idx >= (long)NN * BC_) return;
    int n   = (int)(idx / BC_);
    int col = (int)(idx - (long)n * BC_);
    int b   = col / GIN_;
    int c   = col - b * GIN_;
    long row = (long)b * NN + n;
    float val;
    if (!dec) {
        if (c < D_IN_) val = xsrc[((long)(b * Tt + t) * NN + n) * D_IN_ + c];
        else {
            float hv = h[row * RU_ + (c - D_IN_)];
            val = use_z ? zr[row * (2 * RU_) + (c - D_IN_)] * hv : hv;
        }
    } else {
        if (c == 0)      val = go[row];
        else if (c == 1) val = xsrc[(long)(b * HZN_ + t) * NN + n];
        else {
            float hv = h[row * RU_ + (c - 2)];
            val = use_z ? zr[row * (2 * RU_) + (c - 2)] * hv : hv;
        }
    }
    u16 bv = f2bf(val);
    Xin[(long)n * BC_ + col] = bv;
    XG[row * FPAD_ + c] = bv;                 // identity slot k=0
    XG[row * FPAD_ + 3 * GIN_ + c] = bv;      // identity slot k=3
}

// h = r*h + (1-r)*hc      (r = zr[.., 64:128])
__global__ void hupdate_kernel(const float* __restrict__ zr,
                               const float* __restrict__ hc,
                               float* __restrict__ h) {
    long i = (long)blockIdx.x * blockDim.x + threadIdx.x;
    if (i >= (long)Bb * NN * RU_) return;
    long row = i / RU_;
    int  j   = (int)(i - row * RU_);
    float r  = zr[row * (2 * RU_) + RU_ + j];
    h[i] = r * h[i] + (1.0f - r) * hc[i];
}

// go = h @ proj_W + proj_b ; also write step output
__global__ void proj_kernel(const float* __restrict__ h,
                            const float* __restrict__ projW,
                            const float* __restrict__ projb,
                            float* __restrict__ go,
                            float* __restrict__ outp, int tdec) {
    long row = (long)blockIdx.x * blockDim.x + threadIdx.x;
    if (row >= (long)Bb * NN) return;
    float s = projb[0];
#pragma unroll
    for (int j = 0; j < RU_; ++j) s += h[row * RU_ + j] * projW[j];
    go[row] = s;
    int b = (int)(row >> 11);
    int n = (int)(row & (NN - 1));
    outp[(long)(b * HZN_ + tdec) * NN + n] = s;
}

// ------------------------------------------------------------------
extern "C" void kernel_launch(void* const* d_in, const int* in_sizes, int n_in,
                              void* d_out, int out_size, void* d_ws, size_t ws_size,
                              hipStream_t stream) {
    (void)in_sizes; (void)n_in; (void)out_size; (void)ws_size;
    const float* x      = (const float*)d_in[0];
    const float* ycov   = (const float*)d_in[1];
    const float* adj    = (const float*)d_in[2];
    const float* E1     = (const float*)d_in[3];
    const float* E2     = (const float*)d_in[4];
    const float* egW    = (const float*)d_in[5];
    const float* egb    = (const float*)d_in[6];
    const float* euW    = (const float*)d_in[7];
    const float* eub    = (const float*)d_in[8];
    const float* dgW    = (const float*)d_in[9];
    const float* dgb    = (const float*)d_in[10];
    const float* duW    = (const float*)d_in[11];
    const float* dub    = (const float*)d_in[12];
    const float* projW  = (const float*)d_in[13];
    const float* projb  = (const float*)d_in[14];
    float* outp = (float*)d_out;

    // ---- workspace carve ----
    char* w = (char*)d_ws;
    auto alloc = [&](size_t bytes) { char* p = w; w += (bytes + 255) & ~(size_t)255; return p; };
    u16*   sup  = (u16*)  alloc((size_t)4 * NN * NN * sizeof(u16)); // [0]=adp [1]=cheb(adp) [2]=adj [3]=cheb(adj)
    u16*   Xin  = (u16*)  alloc((size_t)NN * BC_ * sizeof(u16));
    u16*   XG   = (u16*)  alloc((size_t)Bb * NN * FPAD_ * sizeof(u16));
    float* zr   = (float*)alloc((size_t)Bb * NN * 2 * RU_ * sizeof(float));
    float* hc   = (float*)alloc((size_t)Bb * NN * RU_ * sizeof(float));
    float* h    = (float*)alloc((size_t)Bb * NN * RU_ * sizeof(float));
    float* go   = (float*)alloc((size_t)Bb * NN * sizeof(float));
    u16*   Wge  = (u16*)  alloc((size_t)FPAD_ * 128 * sizeof(u16));
    u16*   Wue  = (u16*)  alloc((size_t)FPAD_ * 64  * sizeof(u16));
    u16*   Wgd  = (u16*)  alloc((size_t)FPAD_ * 128 * sizeof(u16));
    u16*   Wud  = (u16*)  alloc((size_t)FPAD_ * 64  * sizeof(u16));

    u16* supA[4] = { sup,
                     sup + (size_t)1 * NN * NN,
                     sup + (size_t)2 * NN * NN,
                     sup + (size_t)3 * NN * NN };
    const int kslotOf[4] = { 1, 2, 4, 5 };   // XG feature slots for the 4 non-identity supports

    // ---- init ----
    hipMemsetAsync(h,  0, (size_t)Bb * NN * RU_ * sizeof(float), stream);
    hipMemsetAsync(go, 0, (size_t)Bb * NN * sizeof(float), stream);
    {
        long n = (long)Bb * NN * (FPAD_ - FREAL);
        zero_pad_kernel<<<(n + 255) / 256, 256, 0, stream>>>(XG);
    }
    adp_softmax_kernel<<<NN, 256, 0, stream>>>(E1, E2, supA[0]);
    {
        long n = (long)NN * NN;
        cast_f32_bf16_kernel<<<(n + 255) / 256, 256, 0, stream>>>(adj, supA[2], n);
    }
    prep_weight_kernel<<<(FPAD_ * 128 + 255) / 256, 256, 0, stream>>>(egW, Wge, FREAL, 128);
    prep_weight_kernel<<<(FPAD_ *  64 + 255) / 256, 256, 0, stream>>>(euW, Wue, FREAL, 64);
    prep_weight_kernel<<<(FPAD_ * 128 + 255) / 256, 256, 0, stream>>>(dgW, Wgd, FREAL, 128);
    prep_weight_kernel<<<(FPAD_ *  64 + 255) / 256, 256, 0, stream>>>(duW, Wud, FREAL, 64);

    // ---- Chebyshev P2 = 2*S@S - I (bf16 WMMA) ----
    dim3 gCheb(NN / BN, NN / BM);
    wmma_gemm_bf16<<<gCheb, 256, 0, stream>>>(supA[0], NN, supA[0], NN, NN,
        supA[1], nullptr, NN, nullptr, MODE_CHEB, 0);
    wmma_gemm_bf16<<<gCheb, 256, 0, stream>>>(supA[2], NN, supA[2], NN, NN,
        supA[3], nullptr, NN, nullptr, MODE_CHEB, 0);

    const dim3 gSup(BC_ / BN, NN / BM);                 // 33 x 16
    const dim3 gGate(128 / BN, (Bb * NN) / BM);         // 2 x 512
    const dim3 gUpd (64  / BN, (Bb * NN) / BM);         // 1 x 512
    const long nPack = (long)NN * BC_;
    const long nHU   = (long)Bb * NN * RU_;

    // ---- one GRU cell: 2 gcns ----
    auto cell = [&](const float* xsrc, const float* goPtr, int t, int dec,
                    const u16* Wg, const float* gb, const u16* Wu, const float* ub) {
        // gate gcn: input = [xt, h]
        pack_kernel<<<(nPack + 255) / 256, 256, 0, stream>>>(
            xsrc, goPtr, h, nullptr, Xin, XG, t, dec, 0);
        for (int s = 0; s < 4; ++s)
            wmma_gemm_bf16<<<gSup, 256, 0, stream>>>(supA[s], NN, Xin, BC_, NN,
                XG, nullptr, 0, nullptr, MODE_SCATTER, kslotOf[s]);
        wmma_gemm_bf16<<<gGate, 256, 0, stream>>>(XG, FPAD_, Wg, 128, FPAD_,
            nullptr, zr, 128, gb, MODE_BIAS_SIG, 0);
        // update gcn: input = [xt, z*h]
        pack_kernel<<<(nPack + 255) / 256, 256, 0, stream>>>(
            xsrc, goPtr, h, zr, Xin, XG, t, dec, 1);
        for (int s = 0; s < 4; ++s)
            wmma_gemm_bf16<<<gSup, 256, 0, stream>>>(supA[s], NN, Xin, BC_, NN,
                XG, nullptr, 0, nullptr, MODE_SCATTER, kslotOf[s]);
        wmma_gemm_bf16<<<gUpd, 256, 0, stream>>>(XG, FPAD_, Wu, 64, FPAD_,
            nullptr, hc, 64, ub, MODE_BIAS_TANH, 0);
        hupdate_kernel<<<(nHU + 255) / 256, 256, 0, stream>>>(zr, hc, h);
    };

    // ---- encoder ----
    for (int t = 0; t < Tt; ++t)
        cell(x, nullptr, t, 0, Wge, egb, Wue, eub);

    // ---- decoder ----
    for (int t = 0; t < HZN_; ++t) {
        cell(ycov, go, t, 1, Wgd, dgb, Wud, dub);
        proj_kernel<<<((long)Bb * NN + 255) / 256, 256, 0, stream>>>(
            h, projW, projb, go, outp, t);
    }
}
